// RAPStrategy_74878459838883
// MI455X (gfx1250) — matrix operand order repaired and checked by reading, not compiled, not run on hardware
//
#include <hip/hip_runtime.h>
#include <hip/hip_bf16.h>

typedef __attribute__((ext_vector_type(16))) __bf16 v16bf;
typedef __attribute__((ext_vector_type(8)))  float  v8f;
typedef unsigned int u32x4 __attribute__((ext_vector_type(4)));
typedef int          i32x4 __attribute__((ext_vector_type(4)));
typedef int          i32x8 __attribute__((ext_vector_type(8)));

#define B_TOT   131072
#define H_DIM   256
#define C_DIM   64
#define K_TOT   320      // H + C concatenated K for stage 1
#define H2_DIM  128
#define O_DIM   128
#define E_NUM   4

#define ROWS    128      // rows per workgroup
#define THREADS 256      // 8 waves of 32
#define KSTEPS1 10       // 320/32
#define NT1     8        // 128/16
#define KSTEPS2 4        // 128/32
#define NT2     8        // 128/16
#define KSTEPSG 8        // 256/32 (gate)
#define SX      328      // Xc LDS row stride in halves (multiple of 8 -> 16B aligned frags)

// workspace layout (uints): per-expert contiguous [W1frag | Wffrag], then Wg frags
#define W1_UINTS_PER_E (KSTEPS1 * NT1 * 32 * 8)     // 20480
#define WF_UINTS_PER_E (KSTEPS2 * NT2 * 32 * 8)     // 8192
#define WE_UINTS       (W1_UINTS_PER_E + WF_UINTS_PER_E)  // 28672 (112 KB)
#define WG_OFF_UINTS   (WE_UINTS * E_NUM)           // 114688
#define WG_UINTS       (KSTEPSG * 32 * 8)           // 2048 (8 KB)
#define WS_TOTAL_UINTS (WG_OFF_UINTS + WG_UINTS)    // 116736

struct Q32 { uint4 a, b; };

static __device__ __forceinline__ unsigned f2bfbits(float f) {
    unsigned u = __builtin_bit_cast(unsigned, f);
    return (u + 0x7FFFu + ((u >> 16) & 1u)) >> 16;   // round-to-nearest-even
}
static __device__ __forceinline__ unsigned pack_bf2(float a, float b) {
    return f2bfbits(a) | (f2bfbits(b) << 16);
}
static __device__ __forceinline__ v8f splat8(float x) {
    v8f r;
#pragma unroll
    for (int i = 0; i < 8; ++i) r[i] = x;
    return r;
}
// A-fragment (16x32 bf16): lane=row (mod 16); halves [0..15] contiguous at p
static __device__ __forceinline__ v16bf ldfragA(const unsigned short* p) {
    Q32 q; q.a = *(const uint4*)p; q.b = *(const uint4*)(p + 8);
    return __builtin_bit_cast(v16bf, q);
}
// B-fragment: 8 consecutive dwords per lane (pre-swizzled in LDS)
static __device__ __forceinline__ v16bf ldfragB(const unsigned* p) {
    Q32 q; q.a = *(const uint4*)p; q.b = *(const uint4*)(p + 4);
    return __builtin_bit_cast(v16bf, q);
}

// ---------------------------------------------------------------------------
// TDM: 1-D tensor load global->LDS, 8-byte elements (bytes % 8 == 0, nelem < 65536).
// D# per CDNA5 ISA ch.8: group0 {count=1, lds_addr, global_addr, type=2},
// group1 {data_size=3 (8B), tensor_dim0=nelem, tensor_dim1=1, tile_dim0=nelem,
//         tile_dim1=1, strides=nelem}.  Tracked by TENSORcnt.
// This toolchain exposes the 6-arg builtin: (g0, g1, g2, g3, g4, cpol).
// ---------------------------------------------------------------------------
static __device__ __forceinline__ void tdm_load_1d(unsigned lds_off, const void* gptr,
                                                   unsigned bytes) {
    unsigned long long ga = (unsigned long long)(uintptr_t)gptr;
    unsigned n = bytes >> 3;                               // 8-byte elements
    u32x4 g0;
    g0[0] = 1u;                                            // count=1 (valid user D#)
    g0[1] = lds_off;                                       // lds_addr (bytes)
    g0[2] = (unsigned)ga;                                  // global_addr[31:0]
    g0[3] = ((unsigned)(ga >> 32) & 0x01FFFFFFu) | (2u << 30);  // addr[56:32] | type=2
    i32x8 g1;
    g1[0] = (int)(3u << 16);                               // wg_mask=0, data_size=3 (8B)
    g1[1] = (int)((n & 0xFFFFu) << 16);                    // tensor_dim0[15:0] @bits63:48
    g1[2] = (int)((n >> 16) | (1u << 16));                 // tensor_dim0[31:16], tensor_dim1.lo=1
    g1[3] = (int)(n << 16);                                // tensor_dim1.hi=0, tile_dim0=n
    g1[4] = 1;                                             // tile_dim1=1, tile_dim2=0
    g1[5] = (int)n;                                        // tensor_dim0_stride[31:0]
    g1[6] = (int)((n & 0xFFFFu) << 16);                    // stride0.hi=0, stride1.lo
    g1[7] = (int)(n >> 16);                                // stride1.hi
    i32x4 z4 = {0, 0, 0, 0};
    i32x8 z8 = {0, 0, 0, 0, 0, 0, 0, 0};
    __builtin_amdgcn_tensor_load_to_lds(g0, g1, z4, z4, z8, 0);
}

// ---------------------------------------------------------------------------
// Prep: fp32 weights -> bf16, pre-swizzled into WMMA B-fragment order.
// uint index ((kt*NT+nt)*32 + lane)*8 + v  holds  (K=kt*32+(lane/16)*16+2v, N=nt*16+lane%16)
// packed as {K low half, K+1 high half}.  Per-expert blocks contiguous for 1-shot TDM.
// ---------------------------------------------------------------------------
__global__ void prep_weights(const float* __restrict__ Ws, const float* __restrict__ Wc,
                             const float* __restrict__ Wf, const float* __restrict__ Wg,
                             unsigned* __restrict__ wsW) {
    int idx = blockIdx.x * blockDim.x + threadIdx.x;
    if (idx >= WS_TOTAL_UINTS) return;
    unsigned val;
    if (idx < WG_OFF_UINTS) {
        int e = idx / WE_UINTS;
        int r = idx % WE_UINTS;
        if (r < W1_UINTS_PER_E) {           // stage-1 weights [Ws;Wc] (K=320)
            int kt = r / (NT1 * 256);
            int r2 = r % (NT1 * 256);
            int nt = r2 / 256;
            int r3 = r2 % 256;
            int l  = r3 / 8, v = r3 % 8;
            int k  = kt * 32 + (l >> 4) * 16 + 2 * v;
            int nn = nt * 16 + (l & 15);
            float f0, f1;
            if (k < H_DIM) {
                const float* p = Ws + ((size_t)e * H_DIM + k) * H2_DIM + nn;
                f0 = p[0]; f1 = p[H2_DIM];
            } else {
                const float* p = Wc + ((size_t)e * C_DIM + (k - H_DIM)) * H2_DIM + nn;
                f0 = p[0]; f1 = p[H2_DIM];
            }
            val = pack_bf2(f0, f1);
        } else {                            // stage-2 weights Wf (K=128)
            int r1 = r - W1_UINTS_PER_E;
            int kt = r1 / (NT2 * 256);
            int r2 = r1 % (NT2 * 256);
            int nt = r2 / 256;
            int r3 = r2 % 256;
            int l  = r3 / 8, v = r3 % 8;
            int k  = kt * 32 + (l >> 4) * 16 + 2 * v;
            int nn = nt * 16 + (l & 15);
            const float* p = Wf + ((size_t)e * H2_DIM + k) * O_DIM + nn;
            val = pack_bf2(p[0], p[O_DIM]);
        }
    } else {                                // gate weights, N padded 4 -> 16 with zeros
        int r  = idx - WG_OFF_UINTS;
        int kt = r / 256;
        int r3 = r % 256;
        int l  = r3 / 8, v = r3 % 8;
        int k  = kt * 32 + (l >> 4) * 16 + 2 * v;
        int nn = l & 15;
        val = (nn < E_NUM) ? pack_bf2(Wg[k * E_NUM + nn], Wg[(k + 1) * E_NUM + nn]) : 0u;
    }
    wsW[idx] = val;
}

// ---------------------------------------------------------------------------
// Main MoE kernel: 128 rows / WG, 8 waves, BF16-WMMA pipeline, TDM weight staging.
// ---------------------------------------------------------------------------
__global__ __launch_bounds__(THREADS) void moe_kernel(
    const float* __restrict__ X,   const float* __restrict__ Ctx,
    const float* __restrict__ bg,
    const float* __restrict__ bs,  const float* __restrict__ bf,
    const unsigned* __restrict__ wsW,
    float* __restrict__ out, float* __restrict__ gout) {

    extern __shared__ char smem[];
    unsigned short* Xc  = (unsigned short*)smem;                        // 128*328 halves  (84 KB)
    unsigned*       W1L = (unsigned*)(smem + ROWS * SX * 2);            // 20480 uints     (80 KB)
    unsigned*       WfL = W1L + W1_UINTS_PER_E;                         //  8192 uints     (32 KB), contiguous after W1L
    unsigned short* Hst = (unsigned short*)(WfL + WF_UINTS_PER_E);      // 128*128 halves  (32 KB)
    float*       gatesL = (float*)(Hst + ROWS * H2_DIM);                // 128*4 floats    ( 2 KB)

    const int tid  = threadIdx.x;
    const int lane = tid & 31;
    const int wave = tid >> 5;
    const int mrow = lane & 15;        // C/D column (mod 16), A row (mod 16)
    const int hs   = lane >> 4;        // lane half-select
    const int rb   = wave * 16;        // wave's row base within tile
    const int row0 = blockIdx.x * ROWS;
    const unsigned w1_lds_off = (unsigned)(uintptr_t)(void*)W1L;

    // Kick off TDM of zero-padded gate-weight fragments into W1L (overlaps X staging)
    if (wave == 0)
        tdm_load_1d(w1_lds_off, wsW + WG_OFF_UINTS, WG_UINTS * 4);

    // ---- Phase 0: stage [X | Ctx] tile into LDS as bf16 --------------------
    for (int p = tid; p < ROWS * (K_TOT / 2); p += THREADS) {
        int row = p / (K_TOT / 2);
        int kk  = (p % (K_TOT / 2)) * 2;
        float2 v;
        if (kk < H_DIM) v = *(const float2*)(X   + (size_t)(row0 + row) * H_DIM + kk);
        else            v = *(const float2*)(Ctx + (size_t)(row0 + row) * C_DIM + (kk - H_DIM));
        *(unsigned*)&Xc[row * SX + kk] = pack_bf2(v.x, v.y);
    }
    if (wave == 0) __builtin_amdgcn_s_wait_tensorcnt(0);
    __syncthreads();

    // ---- Phase 1: gate logits via WMMA (N padded to 16, cols 0..3 valid) ---
    {
        float bgv = bg[mrow & 3];
        v8f gacc = splat8((mrow < E_NUM) ? bgv : 0.0f);
#pragma unroll
        for (int kt = 0; kt < KSTEPSG; ++kt) {
            v16bf a = ldfragA(Xc + (rb + mrow) * SX + kt * 32 + hs * 16);
            v16bf b = ldfragB(W1L + (kt * 32 + lane) * 8);
            gacc = __builtin_amdgcn_wmma_f32_16x16x32_bf16(
                false, a, false, b, (short)0, gacc, false, false);
        }
        if (mrow < E_NUM) {
#pragma unroll
            for (int v = 0; v < 8; ++v)
                gatesL[(rb + v + 8 * hs) * E_NUM + mrow] = gacc[v];
        }
    }
    __syncthreads();
    if (tid < ROWS) {
        float l0 = gatesL[tid * 4 + 0], l1 = gatesL[tid * 4 + 1];
        float l2 = gatesL[tid * 4 + 2], l3 = gatesL[tid * 4 + 3];
        float m  = fmaxf(fmaxf(l0, l1), fmaxf(l2, l3));
        float e0 = __expf(l0 - m), e1 = __expf(l1 - m), e2 = __expf(l2 - m), e3 = __expf(l3 - m);
        float inv = 1.0f / (e0 + e1 + e2 + e3);
        float4 g = make_float4(e0 * inv, e1 * inv, e2 * inv, e3 * inv);
        gatesL[tid * 4 + 0] = g.x; gatesL[tid * 4 + 1] = g.y;
        gatesL[tid * 4 + 2] = g.z; gatesL[tid * 4 + 3] = g.w;
        *(float4*)(gout + (size_t)(row0 + tid) * E_NUM) = g;
    }

    // ---- Phase 2: expert loop ----------------------------------------------
    v8f accF[NT2];
#pragma unroll
    for (int nt = 0; nt < NT2; ++nt) accF[nt] = splat8(0.0f);

    for (int e = 0; e < E_NUM; ++e) {
        __syncthreads();   // prev expert (or gate phase) done reading W1L/WfL
        if (wave == 0) {   // one TDM op stages [W1frag|Wffrag] (112 KB) into LDS
            tdm_load_1d(w1_lds_off, wsW + (size_t)e * WE_UINTS, WE_UINTS * 4);
            __builtin_amdgcn_s_wait_tensorcnt(0);
        }
        __syncthreads();

        // Stage 1: h = relu([x|c] @ W1[e] + bs[e])  (16x320 @ 320x128 per wave)
        {
            v8f acc[NT1];
#pragma unroll
            for (int nt = 0; nt < NT1; ++nt)
                acc[nt] = splat8(bs[e * H2_DIM + nt * 16 + mrow]);
#pragma unroll
            for (int kt = 0; kt < KSTEPS1; ++kt) {
                v16bf a = ldfragA(Xc + (rb + mrow) * SX + kt * 32 + hs * 16);
#pragma unroll
                for (int nt = 0; nt < NT1; ++nt) {
                    v16bf b = ldfragB(W1L + ((kt * NT1 + nt) * 32 + lane) * 8);
                    acc[nt] = __builtin_amdgcn_wmma_f32_16x16x32_bf16(
                        false, a, false, b, (short)0, acc[nt], false, false);
                }
            }
            // ReLU + restage to LDS as bf16 (C-layout -> row-major for A-frags)
#pragma unroll
            for (int nt = 0; nt < NT1; ++nt)
#pragma unroll
                for (int v = 0; v < 8; ++v) {
                    float x = acc[nt][v];
                    x = x > 0.0f ? x : 0.0f;
                    Hst[(rb + v + 8 * hs) * H2_DIM + nt * 16 + mrow] =
                        (unsigned short)f2bfbits(x);
                }
        }
        __syncthreads();

        // Stage 2: oe = h @ Wf[e] + bf[e];  accF += gate[row,e] * oe
        {
            v8f acc2[NT2];
#pragma unroll
            for (int nt = 0; nt < NT2; ++nt)
                acc2[nt] = splat8(bf[e * O_DIM + nt * 16 + mrow]);
#pragma unroll
            for (int kt = 0; kt < KSTEPS2; ++kt) {
                v16bf a = ldfragA(Hst + (rb + mrow) * H2_DIM + kt * 32 + hs * 16);
#pragma unroll
                for (int nt = 0; nt < NT2; ++nt) {
                    v16bf b = ldfragB(WfL + ((kt * NT2 + nt) * 32 + lane) * 8);
                    acc2[nt] = __builtin_amdgcn_wmma_f32_16x16x32_bf16(
                        false, a, false, b, (short)0, acc2[nt], false, false);
                }
            }
#pragma unroll
            for (int v = 0; v < 8; ++v) {
                float g = gatesL[(rb + v + 8 * hs) * E_NUM + e];
#pragma unroll
                for (int nt = 0; nt < NT2; ++nt) accF[nt][v] += g * acc2[nt][v];
            }
        }
    }

    // ---- Phase 3: write final output ---------------------------------------
#pragma unroll
    for (int nt = 0; nt < NT2; ++nt)
#pragma unroll
        for (int v = 0; v < 8; ++v)
            out[(size_t)(row0 + rb + v + 8 * hs) * O_DIM + nt * 16 + mrow] = accF[nt][v];
}

// ---------------------------------------------------------------------------
extern "C" void kernel_launch(void* const* d_in, const int* in_sizes, int n_in,
                              void* d_out, int out_size, void* d_ws, size_t ws_size,
                              hipStream_t stream) {
    (void)in_sizes; (void)n_in; (void)out_size; (void)ws_size;
    const float* X   = (const float*)d_in[0];
    const float* Ctx = (const float*)d_in[1];
    const float* Wg  = (const float*)d_in[2];
    const float* bg  = (const float*)d_in[3];
    const float* Ws  = (const float*)d_in[4];
    const float* Wc  = (const float*)d_in[5];
    const float* bs  = (const float*)d_in[6];
    const float* Wf  = (const float*)d_in[7];
    const float* bf  = (const float*)d_in[8];
    float* out  = (float*)d_out;
    float* gout = out + (size_t)B_TOT * O_DIM;   // outputs concatenated: final, gates
    unsigned* wsW = (unsigned*)d_ws;

    prep_weights<<<(WS_TOTAL_UINTS + 255) / 256, 256, 0, stream>>>(Ws, Wc, Wf, Wg, wsW);

    size_t smem = (size_t)ROWS * SX * 2            // Xc
                + (size_t)W1_UINTS_PER_E * 4       // W1L
                + (size_t)WF_UINTS_PER_E * 4       // WfL
                + (size_t)ROWS * H2_DIM * 2        // Hst
                + (size_t)ROWS * E_NUM * 4;        // gates
    moe_kernel<<<B_TOT / ROWS, THREADS, smem, stream>>>(X, Ctx, bg, bs, bf, wsW, out, gout);
}
